// BotGAT_fmoe_52518860095664
// MI455X (gfx1250) — matrix operand order repaired and checked by reading, not compiled, not run on hardware
//
#include <hip/hip_runtime.h>
#include <hip/hip_bf16.h>

// Problem constants (match reference setup_inputs)
#define NN    100000
#define EE    3200000
#define DD    128
#define DESW  768

typedef __bf16 bf16_t;
typedef __attribute__((ext_vector_type(16))) bf16_t bf16x16;
typedef __attribute__((ext_vector_type(8)))  float  f32x8;
typedef __attribute__((ext_vector_type(4)))  unsigned int u32x4;
typedef __attribute__((ext_vector_type(8)))  int i32x8;
typedef __attribute__((ext_vector_type(4)))  int i32x4;

__device__ __forceinline__ float selu_f(float x) {
  const float l = 1.0507009873554805f, a = 1.6732632423543772f;
  return x > 0.f ? l * x : l * a * (__expf(x) - 1.f);
}
__device__ __forceinline__ float lrelu02(float x) { return x > 0.f ? x : 0.2f * x; }

// ---- TDM: 1-D copy of n_elems bf16 (2B) from global into LDS at byte lds_off.
// D# built per CDNA5 ISA 8.3/8.4: group0 {count=1, lds_addr, global_addr, type=2},
// group1 {data_size=2B, tensor_dim0=tile_dim0=n, tensor_dim1=tile_dim1=1}.
__device__ __forceinline__ void tdm_load_bf16_1d(unsigned lds_off, const void* gptr,
                                                 unsigned n_elems) {
  unsigned long long ga = (unsigned long long)gptr;
  u32x4 g0;
  g0[0] = 1u;                                        // count=1 (valid), user mode
  g0[1] = lds_off;                                   // lds_addr
  g0[2] = (unsigned)ga;                              // global_addr[31:0]
  g0[3] = (unsigned)((ga >> 32) & 0x1FFFFFFu) | (2u << 30);  // addr[56:32] | type=2
  i32x8 g1;
  g1[0] = 0x00010000;                                // data_size=1 (2 bytes)
  g1[1] = (int)((n_elems & 0xFFFFu) << 16);          // tensor_dim0[15:0]
  g1[2] = (int)(((n_elems >> 16) & 0xFFFFu) | (1u << 16)); // dim0[31:16] | tensor_dim1=1
  g1[3] = (int)((n_elems & 0xFFFFu) << 16);          // tile_dim0 (<=65535)
  g1[4] = 1;                                         // tile_dim1=1
  g1[5] = (int)n_elems;                              // tensor_dim0_stride[31:0]
  g1[6] = 0;
  g1[7] = 0;
  i32x4 g2 = {0, 0, 0, 0}, g3 = {0, 0, 0, 0};
#if defined(__clang_major__) && (__clang_major__ >= 23)
  i32x8 g4 = {0, 0, 0, 0, 0, 0, 0, 0};
  __builtin_amdgcn_tensor_load_to_lds(g0, g1, g2, g3, g4, 0);
#else
  __builtin_amdgcn_tensor_load_to_lds(g0, g1, g2, g3, 0);
#endif
}
__device__ __forceinline__ void tdm_wait() {
  __builtin_amdgcn_s_wait_tensorcnt(0);
}

// ---- WMMA fragment builders (wave32; A and B use contiguous k = 16*(lane/16) + s) -------------
// A: 16x32 bf16. lanes 0-15: M=lane, K in {0..15}; lanes 16-31: +16.
__device__ __forceinline__ bf16x16 load_a_frag(const float* X, int row0, int ld,
                                               int k0, int lane) {
  int row = row0 + (lane & 15);
  const float* p = X + (size_t)row * ld + k0 + ((lane >> 4) << 4);
  bf16x16 a;
#pragma unroll
  for (int i = 0; i < 8; ++i) a[i] = (bf16_t)p[i];
#pragma unroll
  for (int i = 0; i < 8; ++i) a[8 + i] = (bf16_t)p[8 + i];
  return a;
}
// B: 32x16 from TRANSPOSED bf16 weights in LDS: Wt[n][k], row length kd.
// lane's 16 K-values are contiguous -> one 32B vector read (ds_load_b128 x2).
__device__ __forceinline__ bf16x16 load_b_frag_t(const bf16_t* Wt, int kd, int k0,
                                                 int n0, int lane) {
  int col = n0 + (lane & 15);
  int kb  = k0 + ((lane >> 4) << 4);
  return *(const bf16x16*)(Wt + (size_t)col * kd + kb);
}
// C/D f32 16x16: lane: N=lane&15; VGPR r: M = r + (lane>=16 ? 8 : 0).

__device__ __forceinline__ f32x8 wmma_bf16(bf16x16 a, bf16x16 b, f32x8 c) {
  return __builtin_amdgcn_wmma_f32_16x16x32_bf16(false, a, false, b, (short)0, c,
                                                 false, false);
}

// ---- Kernel 0: one-time weight convert + transpose to bf16 -----------------
__global__ void prep_weights_kernel(const float* __restrict__ Wd, const float* __restrict__ Wt,
                                    const float* __restrict__ Wi, const float* __restrict__ W1,
                                    const float* __restrict__ W2, const float* __restrict__ We1,
                                    bf16_t* __restrict__ wdT, bf16_t* __restrict__ wtT,
                                    bf16_t* __restrict__ wiT, bf16_t* __restrict__ w1T,
                                    bf16_t* __restrict__ w2T, bf16_t* __restrict__ weT) {
  int i = blockIdx.x * 256 + threadIdx.x;
  if (i < DESW * 32) {                 // 768x32 -> [32][768]
    int k = i >> 5, n = i & 31;
    wdT[n * DESW + k] = (bf16_t)Wd[i];
    wtT[n * DESW + k] = (bf16_t)Wt[i];
  }
  if (i < DD * DD) {                   // 128x128 -> [128][128]^T
    int k = i >> 7, n = i & 127;
    wiT[n * DD + k] = (bf16_t)Wi[i];
    w1T[n * DD + k] = (bf16_t)W1[i];
    w2T[n * DD + k] = (bf16_t)W2[i];
  }
  if (i < 2 * DD * DD) {               // experts: per-expert transpose
    int e = i >> 14, r = i & 16383;
    int k = r >> 7, n = r & 127;
    weT[e * DD * DD + n * DD + k] = (bf16_t)We1[i];
  }
}

// ---- Kernel 1: fused feature encoders + SELU + concat -> x0[N,128] ---------
__global__ void encode_kernel(const float* __restrict__ des, const float* __restrict__ tw,
                              const float* __restrict__ nump, const float* __restrict__ catp,
                              const bf16_t* __restrict__ wdtT,  // wdT||wtT contiguous
                              const float* __restrict__ bd, const float* __restrict__ bt,
                              const float* __restrict__ Wn, const float* __restrict__ bn,
                              const float* __restrict__ Wc, const float* __restrict__ bc,
                              float* __restrict__ x0) {
  extern __shared__ char smem[];
  bf16_t* sWd = (bf16_t*)smem;             // [32][768]
  bf16_t* sWt = sWd + 32 * DESW;           // [32][768]
  int lane = threadIdx.x;
  // TDM: DMA both transposed weight blocks (96KB) into LDS.
  tdm_load_bf16_1d(__builtin_amdgcn_groupstaticsize(), wdtT, 2 * 32 * DESW);
  tdm_wait();
  __syncthreads();

  int row0 = blockIdx.x * 16;
  const f32x8 z8 = {0, 0, 0, 0, 0, 0, 0, 0};
  f32x8 aD0 = z8, aD1 = z8, aT0 = z8, aT1 = z8;
  for (int k0 = 0; k0 < DESW; k0 += 32) {
    bf16x16 a = load_a_frag(des, row0, DESW, k0, lane);
    aD0 = wmma_bf16(a, load_b_frag_t(sWd, DESW, k0, 0, lane), aD0);
    aD1 = wmma_bf16(a, load_b_frag_t(sWd, DESW, k0, 16, lane), aD1);
    a = load_a_frag(tw, row0, DESW, k0, lane);
    aT0 = wmma_bf16(a, load_b_frag_t(sWt, DESW, k0, 0, lane), aT0);
    aT1 = wmma_bf16(a, load_b_frag_t(sWt, DESW, k0, 16, lane), aT1);
  }
  int col = lane & 15, rb = (lane >> 4) << 3;
#pragma unroll
  for (int r = 0; r < 8; ++r) {
    size_t row = (size_t)(row0 + rb + r) * DD;
    x0[row + col]      = selu_f(aD0[r] + bd[col]);
    x0[row + 16 + col] = selu_f(aD1[r] + bd[16 + col]);
    x0[row + 32 + col] = selu_f(aT0[r] + bt[col]);
    x0[row + 48 + col] = selu_f(aT1[r] + bt[16 + col]);
  }
  // num_prop (5->32) and cat_prop (3->32): tiny, VALU
  for (int idx = lane; idx < 16 * 32; idx += 32) {
    int r = idx >> 5, c = idx & 31;
    int row = row0 + r;
    float sn = bn[c];
#pragma unroll
    for (int j = 0; j < 5; ++j) sn += nump[row * 5 + j] * Wn[j * 32 + c];
    float sc = bc[c];
#pragma unroll
    for (int j = 0; j < 3; ++j) sc += catp[row * 3 + j] * Wc[j * 32 + c];
    x0[(size_t)row * DD + 64 + c] = selu_f(sn);
    x0[(size_t)row * DD + 96 + c] = selu_f(sc);
  }
}

// ---- Generic 128x128 GEMM per 16-node tile (WMMA), opt bias / SELU ---------
__global__ void gemm128_kernel(const float* __restrict__ X, const bf16_t* __restrict__ WT,
                               const float* __restrict__ bias, int act,
                               float* __restrict__ Y) {
  extern __shared__ char smem[];
  bf16_t* sW = (bf16_t*)smem;  // transposed [128][128] bf16 = 32KB
  int lane = threadIdx.x;
  tdm_load_bf16_1d(__builtin_amdgcn_groupstaticsize(), WT, DD * DD);
  tdm_wait();
  __syncthreads();

  int row0 = blockIdx.x * 16;
  const f32x8 z8 = {0, 0, 0, 0, 0, 0, 0, 0};
  f32x8 acc[8];
#pragma unroll
  for (int t = 0; t < 8; ++t) acc[t] = z8;
  for (int k0 = 0; k0 < DD; k0 += 32) {
    bf16x16 a = load_a_frag(X, row0, DD, k0, lane);
#pragma unroll
    for (int t = 0; t < 8; ++t)
      acc[t] = wmma_bf16(a, load_b_frag_t(sW, DD, k0, t * 16, lane), acc[t]);
  }
  int col = lane & 15, rb = (lane >> 4) << 3;
#pragma unroll
  for (int t = 0; t < 8; ++t) {
    int c = t * 16 + col;
#pragma unroll
    for (int r = 0; r < 8; ++r) {
      float v = acc[t][r];
      if (bias) v += bias[c];
      if (act == 1) v = selu_f(v);
      Y[(size_t)(row0 + rb + r) * DD + c] = v;
    }
  }
}

// ---- GAT attention-logit dot products: a_s/a_d per (node, head) ------------
__global__ void attn_dots_kernel(const float* __restrict__ xl,
                                 const float* __restrict__ atts,
                                 const float* __restrict__ attd,
                                 float* __restrict__ as_, float* __restrict__ ad_,
                                 int H, int C) {
  int i = blockIdx.x * 256 + threadIdx.x;
  if (i >= NN * H) return;
  int n = i / H, h = i - n * H;
  const float* row = xl + (size_t)n * DD + h * C;
  float s = 0.f, d = 0.f;
  for (int c = 0; c < C; ++c) {
    s += row[c] * atts[h * C + c];
    d += row[c] * attd[h * C + c];
  }
  as_[i] = s;
  ad_[i] = d;
}

// ---- init: agg = bias (broadcast), m = -big, z = 0 -------------------------
__global__ void gat_init_kernel(float* __restrict__ m, float* __restrict__ z,
                                float* __restrict__ agg, const float* __restrict__ bias,
                                int H) {
  int i = blockIdx.x * 256 + threadIdx.x;
  if (i >= NN * DD) return;
  int c = i & (DD - 1), n = i >> 7;
  agg[i] = bias[c];
  if (c < H) { m[n * H + c] = -3.0e38f; z[n * H + c] = 0.f; }
}

__device__ __forceinline__ void atomicMaxF(float* addr, float val) {
  if (val >= 0.f) atomicMax((int*)addr, __float_as_int(val));
  else            atomicMin((unsigned int*)addr, (unsigned int)__float_as_int(val));
}

// ---- edge pass 1: segment max of leaky_relu(a_s[s]+a_d[d]) -----------------
__global__ void edge_max_kernel(const int* __restrict__ src, const int* __restrict__ dst,
                                const float* __restrict__ as_, const float* __restrict__ ad_,
                                float* __restrict__ m, int H) {
  int i = blockIdx.x * 256 + threadIdx.x;
  if (i >= (EE + NN) * H) return;
  int e = i / H, h = i - e * H;
  int s = e < EE ? src[e] : e - EE;
  int d = e < EE ? dst[e] : e - EE;
  float al = lrelu02(as_[s * H + h] + ad_[d * H + h]);
  atomicMaxF(&m[d * H + h], al);
}

// ---- edge pass 2: z[d] += exp(alpha - m[d]); prefetch xl[s] for pass 3 -----
__global__ void edge_sum_kernel(const int* __restrict__ src, const int* __restrict__ dst,
                                const float* __restrict__ as_, const float* __restrict__ ad_,
                                const float* __restrict__ m, float* __restrict__ z,
                                const float* __restrict__ xl, int H) {
  int i = blockIdx.x * 256 + threadIdx.x;
  if (i >= (EE + NN) * H) return;
  int e = i / H, h = i - e * H;
  int s = e < EE ? src[e] : e - EE;
  int d = e < EE ? dst[e] : e - EE;
  float al = lrelu02(as_[s * H + h] + ad_[d * H + h]);
  atomicAdd(&z[d * H + h], __expf(al - m[d * H + h]));
  __builtin_prefetch(xl + (size_t)s * DD, 0, 1);  // warm L2 for the agg pass
}

// ---- edge pass 3: agg[d] += a * xl[s]; one wave-lane per 4 channels --------
__global__ void edge_agg_kernel(const int* __restrict__ src, const int* __restrict__ dst,
                                const float* __restrict__ as_, const float* __restrict__ ad_,
                                const float* __restrict__ m, const float* __restrict__ z,
                                const float* __restrict__ xl, float* __restrict__ agg,
                                int H) {
  unsigned int tid = blockIdx.x * 256 + threadIdx.x;
  if (tid >= (unsigned int)(EE + NN) * 32u) return;
  int e  = tid >> 5;
  int c0 = (tid & 31) * 4;
  int h  = (c0 * H) >> 7;  // C = 128/H
  int s = e < EE ? src[e] : e - EE;
  int d = e < EE ? dst[e] : e - EE;
  float al = lrelu02(as_[s * H + h] + ad_[d * H + h]);
  float a  = __expf(al - m[d * H + h]) / z[d * H + h];
  const float4 v = *(const float4*)(xl + (size_t)s * DD + c0);
  float* op = agg + (size_t)d * DD + c0;
  atomicAdd(op + 0, a * v.x);
  atomicAdd(op + 1, a * v.y);
  atomicAdd(op + 2, a * v.z);
  atomicAdd(op + 3, a * v.w);
}

// ---- MoE: top-1 of 2 experts; both expert MLPs via WMMA on 16-node tile ----
__global__ void moe_kernel(const float* __restrict__ X, const float* __restrict__ Wg,
                           const bf16_t* __restrict__ weT, const float* __restrict__ be1,
                           const float* __restrict__ We2, const float* __restrict__ be2,
                           float* __restrict__ out) {
  extern __shared__ char smem[];
  bf16_t* sW = (bf16_t*)smem;                        // 2 x [128][128]^T bf16 = 64KB
  float*  sX = (float*)(smem + 2 * DD * DD * 2);     // 16*128 f32
  float*  sH = sX + 16 * DD;                         // 2*16*128 f32
  int lane = threadIdx.x;
  int row0 = blockIdx.x * 16;
  tdm_load_bf16_1d(__builtin_amdgcn_groupstaticsize(), weT, 2 * DD * DD);
  for (int i = lane; i < 16 * DD; i += 32) sX[i] = X[(size_t)row0 * DD + i];
  tdm_wait();
  __syncthreads();

  const f32x8 z8 = {0, 0, 0, 0, 0, 0, 0, 0};
  int col = lane & 15, rb = (lane >> 4) << 3;
  for (int ex = 0; ex < 2; ++ex) {
    f32x8 acc[8];
#pragma unroll
    for (int t = 0; t < 8; ++t) acc[t] = z8;
    const bf16_t* We = sW + ex * DD * DD;
    for (int k0 = 0; k0 < DD; k0 += 32) {
      bf16x16 a = load_a_frag(sX, 0, DD, k0, lane);
#pragma unroll
      for (int t = 0; t < 8; ++t)
        acc[t] = wmma_bf16(a, load_b_frag_t(We, DD, k0, t * 16, lane), acc[t]);
    }
#pragma unroll
    for (int t = 0; t < 8; ++t) {
      int c = t * 16 + col;
#pragma unroll
      for (int r = 0; r < 8; ++r) {
        float v = acc[t][r] + be1[ex * DD + c];
        v = v > 0.f ? v : 0.01f * v;   // leaky_relu(0.01)
        sH[ex * 16 * DD + (rb + r) * DD + c] = v;
      }
    }
  }
  __syncthreads();
  if (lane < 16) {
    int n = row0 + lane;
    const float* xr = sX + lane * DD;
    float g0 = 0.f, g1 = 0.f;
    for (int k = 0; k < DD; ++k) { g0 += xr[k] * Wg[k * 2]; g1 += xr[k] * Wg[k * 2 + 1]; }
    int ex = (g1 > g0) ? 1 : 0;       // top-1 gate; weight renormalizes to 1
    const float* hr = sH + ex * 16 * DD + lane * DD;
    float o0 = be2[ex * 2], o1 = be2[ex * 2 + 1];
    for (int k = 0; k < DD; ++k) {
      o0 += hr[k] * We2[ex * 256 + k * 2];
      o1 += hr[k] * We2[ex * 256 + k * 2 + 1];
    }
    out[(size_t)n * 2]     = o0;
    out[(size_t)n * 2 + 1] = o1;
  }
}

extern "C" void kernel_launch(void* const* d_in, const int* in_sizes, int n_in,
                              void* d_out, int out_size, void* d_ws, size_t ws_size,
                              hipStream_t stream) {
  const float* des   = (const float*)d_in[0];
  const float* tweet = (const float*)d_in[1];
  const float* nump  = (const float*)d_in[2];
  const float* catp  = (const float*)d_in[3];
  const int*   eidx  = (const int*)d_in[4];
  // d_in[5] = edge_type (unused by forward)
  const float* Wd = (const float*)d_in[6];  const float* bd = (const float*)d_in[7];
  const float* Wt = (const float*)d_in[8];  const float* bt = (const float*)d_in[9];
  const float* Wn = (const float*)d_in[10]; const float* bn = (const float*)d_in[11];
  const float* Wc = (const float*)d_in[12]; const float* bc = (const float*)d_in[13];
  const float* Wi = (const float*)d_in[14]; const float* bi = (const float*)d_in[15];
  const float* W1 = (const float*)d_in[16];
  const float* a1s = (const float*)d_in[17]; const float* a1d = (const float*)d_in[18];
  const float* b1 = (const float*)d_in[19];
  const float* W2 = (const float*)d_in[20];
  const float* a2s = (const float*)d_in[21]; const float* a2d = (const float*)d_in[22];
  const float* b2 = (const float*)d_in[23];
  const float* Wg  = (const float*)d_in[24];
  const float* We1 = (const float*)d_in[25]; const float* be1 = (const float*)d_in[26];
  const float* We2 = (const float*)d_in[27]; const float* be2 = (const float*)d_in[28];
  float* out = (float*)d_out;

  const int* srcI = eidx;
  const int* dstI = eidx + EE;

  // workspace layout: f32 activations/stats, then bf16 transposed weights
  float* p    = (float*)d_ws;
  float* bufA = p;                         // N*128
  float* bufB = bufA + (size_t)NN * DD;    // N*128
  float* bufC = bufB + (size_t)NN * DD;    // N*128
  float* as1  = bufC + (size_t)NN * DD;    // N*4
  float* ad1  = as1 + (size_t)NN * 4;
  float* m1   = ad1 + (size_t)NN * 4;
  float* z1   = m1 + (size_t)NN * 4;
  float* as2  = z1 + (size_t)NN * 4;       // N
  float* ad2  = as2 + NN;
  float* m2   = ad2 + NN;
  float* z2   = m2 + NN;
  bf16_t* wdT = (bf16_t*)(z2 + NN);        // [32][768]
  bf16_t* wtT = wdT + 32 * DESW;           // [32][768] (contiguous with wdT)
  bf16_t* wiT = wtT + 32 * DESW;           // [128][128]
  bf16_t* w1T = wiT + DD * DD;
  bf16_t* w2T = w1T + DD * DD;
  bf16_t* weT = w2T + DD * DD;             // [2][128][128]

  const int NB16   = NN / 16;                        // 6250
  const int gInit  = (NN * DD) / 256;                // 50000
  const int gDot4  = (NN * 4 + 255) / 256;
  const int gDot1  = (NN + 255) / 256;
  const int gE4    = ((EE + NN) * 4 + 255) / 256;
  const int gE1    = ((EE + NN) * 1 + 255) / 256;
  const int gAgg   = ((EE + NN) * 32) / 256;         // 412500
  const int gPrep  = (2 * DD * DD + 255) / 256;      // 128

  // 0) one-time: bf16 transposed weights
  prep_weights_kernel<<<gPrep, 256, 0, stream>>>(Wd, Wt, Wi, W1, W2, We1,
                                                 wdT, wtT, wiT, w1T, w2T, weT);
  // 1) feature encoders -> x0 (bufA)
  encode_kernel<<<NB16, 32, 2 * 32 * DESW * sizeof(bf16_t), stream>>>(
      des, tweet, nump, catp, wdT, bd, bt, Wn, bn, Wc, bc, bufA);
  // 2) x1 = selu(x0 @ Wi + bi) -> bufB
  gemm128_kernel<<<NB16, 32, DD * DD * sizeof(bf16_t), stream>>>(bufA, wiT, bi, 1, bufB);
  // 3) GAT1 pre: xl1 = x1 @ W1 -> bufA ; attention dots
  gemm128_kernel<<<NB16, 32, DD * DD * sizeof(bf16_t), stream>>>(bufB, w1T, nullptr, 0, bufA);
  attn_dots_kernel<<<gDot4, 256, 0, stream>>>(bufA, a1s, a1d, as1, ad1, 4, 32);
  gat_init_kernel<<<gInit, 256, 0, stream>>>(m1, z1, bufC, b1, 4);
  edge_max_kernel<<<gE4, 256, 0, stream>>>(srcI, dstI, as1, ad1, m1, 4);
  edge_sum_kernel<<<gE4, 256, 0, stream>>>(srcI, dstI, as1, ad1, m1, z1, bufA, 4);
  edge_agg_kernel<<<gAgg, 256, 0, stream>>>(srcI, dstI, as1, ad1, m1, z1, bufA, bufC, 4);
  // 4) GAT2 pre: xl2 = h1 @ W2 -> bufB ; dots
  gemm128_kernel<<<NB16, 32, DD * DD * sizeof(bf16_t), stream>>>(bufC, w2T, nullptr, 0, bufB);
  attn_dots_kernel<<<gDot1, 256, 0, stream>>>(bufB, a2s, a2d, as2, ad2, 1, 128);
  gat_init_kernel<<<gInit, 256, 0, stream>>>(m2, z2, bufA, b2, 1);
  edge_max_kernel<<<gE1, 256, 0, stream>>>(srcI, dstI, as2, ad2, m2, 1);
  edge_sum_kernel<<<gE1, 256, 0, stream>>>(srcI, dstI, as2, ad2, m2, z2, bufB, 1);
  edge_agg_kernel<<<gAgg, 256, 0, stream>>>(srcI, dstI, as2, ad2, m2, z2, bufB, bufA, 1);
  // 5) MoE head -> out
  moe_kernel<<<NB16, 32, 2 * DD * DD * 2 + 3 * 16 * DD * (int)sizeof(float), stream>>>(
      bufA, Wg, weT, be1, We2, be2, out);
}
